// SelfAttention_52767968198794
// MI455X (gfx1250) — matrix operand order repaired and checked by reading, not compiled, not run on hardware
//
#include <hip/hip_runtime.h>
#include <hip/hip_bf16.h>

// ---------------------------------------------------------------------------
// Self-attention on MI455X (gfx1250), bf16 WMMA pipeline.
// B=16, N=D=1024.
//   vk/vq/vv = x^T @ W{k,q,v}            (bf16 in, f32 acc, bf16 out)
//   kq       = vk @ vq^T / 32            (bf16 in, f32 acc, bf16 out, TDM-staged)
//   sm       = softmax(kq, axis=-1)      (f32 math, bf16 in/out, in place)
//   out[b,n,d] = (sm @ vv)[b,d,n]        (bf16 in, f32 acc, f32 transposed out)
// ---------------------------------------------------------------------------

typedef __attribute__((ext_vector_type(16))) __bf16        v16bf;
typedef __attribute__((ext_vector_type(8)))  float         v8f;
typedef __attribute__((ext_vector_type(4)))  unsigned int  v4u;
typedef __attribute__((ext_vector_type(8)))  int           v8i;
typedef __attribute__((ext_vector_type(4)))  int           v4i;

#define TILE 128
#define KT   32
// LDS row stride in halfwords: 40 -> 80B rows. 16 lanes * 20-bank stride mod 64
// hits 16 distinct banks (conflict-free pair loads) AND rows stay 16B-aligned
// so row-major tiles can be staged with ds_store_b128 / TDM.
#define LDK  40

#ifndef __has_builtin
#define __has_builtin(x) 0
#endif
#if __has_builtin(__builtin_amdgcn_tensor_load_to_lds)
#define USE_TDM 1
#else
#define USE_TDM 0
#endif

static __device__ __forceinline__ unsigned short f32_to_bf16(float f) {
    unsigned int u = __float_as_uint(f);
    u += 0x7FFFu + ((u >> 16) & 1u);      // round to nearest even
    return (unsigned short)(u >> 16);
}
static __device__ __forceinline__ float bf16_to_f32(unsigned short h) {
    return __uint_as_float(((unsigned int)h) << 16);
}

// A fragment (16x32 bf16, M x K). base -> element [row0][0], row stride LDK.
// ISA 7.12.2: lanes 0-15 M=lane, lanes 16-31 M=lane-16;
// VGPR q holds K = (q<4 ? 2q : 2q+8) + 8*(lane>>4), and K+1.
static __device__ __forceinline__ v16bf frag_a_lds(const unsigned short* base, int lane) {
    union { v16bf v; unsigned int u[8]; } r;
    const int half = lane >> 4;
    const int m    = lane & 15;
    const unsigned short* p = base + m * LDK;
#pragma unroll
    for (int q = 0; q < 8; ++q) {
        int k = ((q < 4) ? (2 * q) : (2 * q + 8)) + 8 * half;
        r.u[q] = *(const unsigned int*)(p + k);
    }
    return r.v;
}

// B fragment (32x16 bf16, K x N) from LDS stored as Bt[n][k].
// Same K order as A: lanes 0-15 N=lane, lanes 16-31 N=lane-16;
// VGPR q holds K = (q<4 ? 2q : 2q+8) + 8*(lane>>4), and K+1.
static __device__ __forceinline__ v16bf frag_b_lds(const unsigned short* base, int lane) {
    union { v16bf v; unsigned int u[8]; } r;
    const int half = lane >> 4;
    const int n    = lane & 15;
    const unsigned short* p = base + n * LDK;
#pragma unroll
    for (int q = 0; q < 8; ++q) {
        int k = ((q < 4) ? (2 * q) : (2 * q + 8)) + 8 * half;
        r.u[q] = *(const unsigned int*)(p + k);
    }
    return r.v;
}

static __device__ __forceinline__ v8f wmma_bf16(v16bf a, v16bf b, v8f c) {
    return __builtin_amdgcn_wmma_f32_16x16x32_bf16(
        /*neg_a=*/false, a, /*neg_b=*/false, b,
        /*c_mod=*/(short)0, c, /*reuse_a=*/false, /*reuse_b=*/false);
}

// Stage a row-major 128x32 bf16 tile (row stride row_stride elems in global)
// into LDS with b128 loads + b128 stores. dst rows are 80B (16B-aligned).
static __device__ __forceinline__ void stage_rowmajor_tile(
    unsigned short (*dst)[LDK], const unsigned short* __restrict__ src,
    size_t row_stride, int t) {
#pragma unroll
    for (int i = 0; i < 2; ++i) {
        int s   = t + 256 * i;        // 512 segments of 8 bf16
        int row = s >> 2;             // 128 rows
        int seg = s & 3;              // 4 x 16B per 64B row
        uint4 v = *(const uint4*)(src + (size_t)row * row_stride + seg * 8);
        *(uint4*)&dst[row][seg * 8] = v;
    }
}

#if USE_TDM
// ---------------------------------------------------------------------------
// Tensor Data Mover: load a 2-D tile (tile_dim0=32 bf16 contiguous,
// tile_dim1=128 rows, tensor_dim0_stride=1024) into LDS at lds_off,
// padding each 16-dword (64B) row with 4 dwords (16B) -> LDK=40 halfword rows.
// D# built per CDNA5 ISA §8.3 (group 0) / §8.4 (group 1).
// ---------------------------------------------------------------------------
static __device__ __forceinline__ void tdm_load_tile(unsigned lds_off,
                                                     const unsigned short* gptr) {
    unsigned long long ga = (unsigned long long)(size_t)gptr;
    v4u g0;
    g0[0] = 1u;                                    // count=1, user descriptor
    g0[1] = lds_off;                               // lds_addr (bytes)
    g0[2] = (unsigned)(ga & 0xFFFFFFFFu);          // global_addr[31:0]
    g0[3] = (unsigned)((ga >> 32) & 0x01FFFFFFull) // global_addr[56:32]
          | (2u << 30);                            // type=2 ("image")
    v8i g1;
    g1[0] = (int)((1u << 16)     // data_size = 1 -> 2 bytes
                | (1u << 20)     // pad_enable
                | (3u << 22)     // pad_interval = 16 dwords (64B)
                | (3u << 25));   // pad_amount   = 4 dwords (16B)
    g1[1] = (int)(1024u << 16);  // tensor_dim0[15:0]  @ bits 63:48
    g1[2] = (int)(1024u << 16);  // tensor_dim0 hi = 0; tensor_dim1[15:0] @ 95:80
    g1[3] = (int)(32u << 16);    // tensor_dim1 hi = 0; tile_dim0=32 @ 127:112
    g1[4] = 128;                 // tile_dim1 = 128; tile_dim2 = 0
    g1[5] = 1024;                // tensor_dim0_stride[31:0]
    g1[6] = 0;                   // stride hi / tensor_dim1_stride lo
    g1[7] = 0;
    v4i gz;
    gz[0] = 0; gz[1] = 0; gz[2] = 0; gz[3] = 0;
#if __clang_major__ <= 22
    __builtin_amdgcn_tensor_load_to_lds(g0, g1, gz, gz, 0);
#else
    v8i g4;
    g4[0] = 0; g4[1] = 0; g4[2] = 0; g4[3] = 0;
    g4[4] = 0; g4[5] = 0; g4[6] = 0; g4[7] = 0;
    __builtin_amdgcn_tensor_load_to_lds(g0, g1, gz, gz, g4, 0);
#endif
}
#endif  // USE_TDM

// ---------------------------------------------------------------------------
// Kernel 0: f32 -> bf16 convert (weights)
// ---------------------------------------------------------------------------
__global__ void cvt_bf16_kernel(const float* __restrict__ in,
                                unsigned short* __restrict__ out, int n) {
    int i = blockIdx.x * blockDim.x + threadIdx.x;
    int stride = gridDim.x * blockDim.x;
    for (; i < n; i += stride) out[i] = f32_to_bf16(in[i]);
}

// ---------------------------------------------------------------------------
// Kernel 1: vk/vq/vv[b,d,n] = sum_m x[b,m,d] * W[m,n]   (fused transpose of x)
// grid (8, 8, 48): x=n-tile, y=d-tile, z = b*3 + which
// ---------------------------------------------------------------------------
__global__ __launch_bounds__(256) void qkv_gemm_kernel(
    const float* __restrict__ x,
    const unsigned short* __restrict__ wk,
    const unsigned short* __restrict__ wq,
    const unsigned short* __restrict__ wv,
    unsigned short* __restrict__ vk,
    unsigned short* __restrict__ vq,
    unsigned short* __restrict__ vv) {

    const int which = blockIdx.z % 3;
    const int b     = blockIdx.z / 3;
    const unsigned short* __restrict__ w =
        (which == 0) ? wk : (which == 1) ? wq : wv;
    unsigned short* __restrict__ out =
        (which == 0) ? vk : (which == 1) ? vq : vv;

    const int n0 = blockIdx.x * TILE;
    const int d0 = blockIdx.y * TILE;

    __shared__ unsigned short As[TILE][LDK];  // As[d][k], k = m-local
    __shared__ unsigned short Bs[TILE][LDK];  // Bs[n][k]

    const int t    = threadIdx.x;
    const int lane = t & 31;
    const int wid  = t >> 5;
    const int wm   = wid & 3;    // 4 row blocks of 32 (d)
    const int wn   = wid >> 2;   // 2 col blocks of 64 (n)

    v8f acc[2][4];
#pragma unroll
    for (int i = 0; i < 2; ++i)
#pragma unroll
        for (int j = 0; j < 4; ++j)
#pragma unroll
            for (int r = 0; r < 8; ++r) acc[i][j][r] = 0.f;

    for (int k0 = 0; k0 < 1024; k0 += KT) {
        const float* xrow          = x + (size_t)b * 1024 * 1024 + (size_t)k0 * 1024 + d0;
        const unsigned short* wrow = w + (size_t)k0 * 1024 + n0;

        // A tile: As[d][k] = bf16(x[b][k0+k][d0+d]); b128 loads, b16 scatter.
#pragma unroll
        for (int i = 0; i < 4; ++i) {
            int s   = t + 256 * i;    // 1024 segments of 4 floats
            int k   = s >> 5;         // 32 k-rows
            int seg = s & 31;         // 32 x float4 per 512B row
            float4 f = *(const float4*)(xrow + (size_t)k * 1024 + seg * 4);
            int d = seg * 4;
            As[d + 0][k] = f32_to_bf16(f.x);
            As[d + 1][k] = f32_to_bf16(f.y);
            As[d + 2][k] = f32_to_bf16(f.z);
            As[d + 3][k] = f32_to_bf16(f.w);
        }
        // B tile (transposed): Bs[n][k] = W[k0+k][n0+n]; b128 loads, b16 scatter.
#pragma unroll
        for (int i = 0; i < 2; ++i) {
            int s  = t + 256 * i;     // 512 segments of 8 bf16
            int k  = s >> 4;          // 32 k-rows
            int cg = s & 15;          // 16 x 8-col groups
            uint4 vv4 = *(const uint4*)(wrow + (size_t)k * 1024 + cg * 8);
            const unsigned short* e = (const unsigned short*)&vv4;
#pragma unroll
            for (int j = 0; j < 8; ++j) Bs[cg * 8 + j][k] = e[j];
        }
        if (k0 + KT < 1024) {  // pull next tiles toward GL2
            __builtin_prefetch(xrow + (size_t)KT * 1024 + (t & 31) * 32, 0, 1);
            __builtin_prefetch(wrow + (size_t)KT * 1024 + (t & 31) * 32, 0, 1);
        }
        __syncthreads();

        v16bf afrag[2], bfrag[4];
#pragma unroll
        for (int i = 0; i < 2; ++i)
            afrag[i] = frag_a_lds(&As[wm * 32 + 16 * i][0], lane);
#pragma unroll
        for (int j = 0; j < 4; ++j)
            bfrag[j] = frag_b_lds(&Bs[wn * 64 + 16 * j][0], lane);
#pragma unroll
        for (int i = 0; i < 2; ++i)
#pragma unroll
            for (int j = 0; j < 4; ++j)
                acc[i][j] = wmma_bf16(afrag[i], bfrag[j], acc[i][j]);

        __syncthreads();
    }

    // Epilogue: C layout -> bf16 store. m = 8*(lane>>4)+r, n = lane&15.
    const int half = lane >> 4;
    const int cn   = lane & 15;
#pragma unroll
    for (int i = 0; i < 2; ++i)
#pragma unroll
        for (int j = 0; j < 4; ++j)
#pragma unroll
            for (int r = 0; r < 8; ++r) {
                int d  = d0 + wm * 32 + 16 * i + 8 * half + r;
                int nn = n0 + wn * 64 + 16 * j + cn;
                out[((size_t)b * 1024 + d) * 1024 + nn] = f32_to_bf16(acc[i][j][r]);
            }
}

// ---------------------------------------------------------------------------
// Kernel 2: kq[b,d,e] = (sum_n vk[b,d,n]*vq[b,e,n]) / 32   -> bf16
// Both tiles are row-major over the contraction axis -> staged by the
// Tensor Data Mover (with HW LDS padding to LDK rows) when available.
// grid (8, 8, 16): x=e-tile, y=d-tile, z=b
// ---------------------------------------------------------------------------
__global__ __launch_bounds__(256) void kq_gemm_kernel(
    const unsigned short* __restrict__ vk,
    const unsigned short* __restrict__ vq,
    unsigned short* __restrict__ kq) {

    const int b  = blockIdx.z;
    const int e0 = blockIdx.x * TILE;
    const int d0 = blockIdx.y * TILE;

    __shared__ unsigned short smem[2][TILE][LDK];
    unsigned short (*As)[LDK] = smem[0];  // As[d][k], k = n-local
    unsigned short (*Bs)[LDK] = smem[1];  // Bs[e][k]

    const int t    = threadIdx.x;
    const int lane = t & 31;
    const int wid  = t >> 5;
    const int wm   = wid & 3;
    const int wn   = wid >> 2;

    v8f acc[2][4];
#pragma unroll
    for (int i = 0; i < 2; ++i)
#pragma unroll
        for (int j = 0; j < 4; ++j)
#pragma unroll
            for (int r = 0; r < 8; ++r) acc[i][j][r] = 0.f;

    for (int k0 = 0; k0 < 1024; k0 += KT) {
        const unsigned short* asrc = vk + ((size_t)b * 1024 + d0) * 1024 + k0;
        const unsigned short* bsrc = vq + ((size_t)b * 1024 + e0) * 1024 + k0;
#if USE_TDM
        if (wid == 0) {  // TDM ignores EXEC; issue once from wave 0
            tdm_load_tile(0u, asrc);
            tdm_load_tile((unsigned)(TILE * LDK * 2), bsrc);
            __builtin_amdgcn_s_wait_tensorcnt(0);
        }
        __syncthreads();
#else
        stage_rowmajor_tile(As, asrc, 1024, t);
        stage_rowmajor_tile(Bs, bsrc, 1024, t);
        __syncthreads();
#endif

        v16bf afrag[2], bfrag[4];
#pragma unroll
        for (int i = 0; i < 2; ++i)
            afrag[i] = frag_a_lds(&As[wm * 32 + 16 * i][0], lane);
#pragma unroll
        for (int j = 0; j < 4; ++j)
            bfrag[j] = frag_b_lds(&Bs[wn * 64 + 16 * j][0], lane);
#pragma unroll
        for (int i = 0; i < 2; ++i)
#pragma unroll
            for (int j = 0; j < 4; ++j)
                acc[i][j] = wmma_bf16(afrag[i], bfrag[j], acc[i][j]);

        __syncthreads();
    }

    const int half = lane >> 4;
    const int cn   = lane & 15;
#pragma unroll
    for (int i = 0; i < 2; ++i)
#pragma unroll
        for (int j = 0; j < 4; ++j)
#pragma unroll
            for (int r = 0; r < 8; ++r) {
                int d  = d0 + wm * 32 + 16 * i + 8 * half + r;
                int ee = e0 + wn * 64 + 16 * j + cn;
                kq[((size_t)b * 1024 + d) * 1024 + ee] =
                    f32_to_bf16(acc[i][j][r] * 0.03125f);  // 1/sqrt(1024)
            }
}

// ---------------------------------------------------------------------------
// Kernel 3: in-place row softmax over last axis (rows of 1024 bf16)
// grid: 16*1024 blocks of 256 threads, one block per (b,d) row
// ---------------------------------------------------------------------------
__global__ __launch_bounds__(256) void softmax_kernel(unsigned short* __restrict__ kq) {
    unsigned short* p = kq + (size_t)blockIdx.x * 1024;
    __shared__ float red[256];
    const int t = threadIdx.x;

    float v[4];
    float m = -3.402823466e38f;
#pragma unroll
    for (int i = 0; i < 4; ++i) {
        v[i] = bf16_to_f32(p[t + 256 * i]);
        m = fmaxf(m, v[i]);
    }
    red[t] = m;
    __syncthreads();
    for (int s = 128; s > 0; s >>= 1) {
        if (t < s) red[t] = fmaxf(red[t], red[t + s]);
        __syncthreads();
    }
    m = red[0];
    __syncthreads();

    float sum = 0.f;
#pragma unroll
    for (int i = 0; i < 4; ++i) {
        v[i] = __expf(v[i] - m);
        sum += v[i];
    }
    red[t] = sum;
    __syncthreads();
    for (int s = 128; s > 0; s >>= 1) {
        if (t < s) red[t] += red[t + s];
        __syncthreads();
    }
    float inv = 1.f / red[0];
#pragma unroll
    for (int i = 0; i < 4; ++i) p[t + 256 * i] = f32_to_bf16(v[i] * inv);
}

// ---------------------------------------------------------------------------
// Kernel 4: out[b,n,d] = sum_e sm[b,d,e] * vv[b,e,n]   (f32, transposed store)
// grid (8, 8, 16): x=n-tile, y=d-tile, z=b
// ---------------------------------------------------------------------------
__global__ __launch_bounds__(256) void out_gemm_kernel(
    const unsigned short* __restrict__ sm,
    const unsigned short* __restrict__ vv,
    float* __restrict__ out) {

    const int b  = blockIdx.z;
    const int n0 = blockIdx.x * TILE;
    const int d0 = blockIdx.y * TILE;

    __shared__ unsigned short As[TILE][LDK];  // As[d][k], k = e-local
    __shared__ unsigned short Bs[TILE][LDK];  // Bs[n][k]

    const int t    = threadIdx.x;
    const int lane = t & 31;
    const int wid  = t >> 5;
    const int wm   = wid & 3;
    const int wn   = wid >> 2;

    v8f acc[2][4];
#pragma unroll
    for (int i = 0; i < 2; ++i)
#pragma unroll
        for (int j = 0; j < 4; ++j)
#pragma unroll
            for (int r = 0; r < 8; ++r) acc[i][j][r] = 0.f;

    for (int k0 = 0; k0 < 1024; k0 += KT) {
        // A: sm rows d, contiguous over e -> vectorized row-major staging.
        stage_rowmajor_tile(As, sm + ((size_t)b * 1024 + d0) * 1024 + k0, 1024, t);

        // B (transposed): Bs[n][k] = vv[b][k0+k][n0+n]; b128 loads, b16 scatter.
        const unsigned short* vrow = vv + ((size_t)b * 1024 + k0) * 1024 + n0;
#pragma unroll
        for (int i = 0; i < 2; ++i) {
            int s  = t + 256 * i;
            int k  = s >> 4;
            int cg = s & 15;
            uint4 vv4 = *(const uint4*)(vrow + (size_t)k * 1024 + cg * 8);
            const unsigned short* e = (const unsigned short*)&vv4;
#pragma unroll
            for (int j = 0; j < 8; ++j) Bs[cg * 8 + j][k] = e[j];
        }
        if (k0 + KT < 1024) {
            __builtin_prefetch(vrow + (size_t)KT * 1024 + (t & 31) * 32, 0, 1);
        }
        __syncthreads();

        v16bf afrag[2], bfrag[4];
#pragma unroll
        for (int i = 0; i < 2; ++i)
            afrag[i] = frag_a_lds(&As[wm * 32 + 16 * i][0], lane);
#pragma unroll
        for (int j = 0; j < 4; ++j)
            bfrag[j] = frag_b_lds(&Bs[wn * 64 + 16 * j][0], lane);
#pragma unroll
        for (int i = 0; i < 2; ++i)
#pragma unroll
            for (int j = 0; j < 4; ++j)
                acc[i][j] = wmma_bf16(afrag[i], bfrag[j], acc[i][j]);

        __syncthreads();
    }

    // Transposed f32 store: final output is (B, N, D) = out[b][nn][d]
    const int half = lane >> 4;
    const int cn   = lane & 15;
#pragma unroll
    for (int i = 0; i < 2; ++i)
#pragma unroll
        for (int j = 0; j < 4; ++j)
#pragma unroll
            for (int r = 0; r < 8; ++r) {
                int d  = d0 + wm * 32 + 16 * i + 8 * half + r;
                int nn = n0 + wn * 64 + 16 * j + cn;
                out[((size_t)b * 1024 + nn) * 1024 + d] = acc[i][j][r];
            }
}

// ---------------------------------------------------------------------------
// Host launcher. Workspace layout (bytes):
//   [0,2M)    wk16   [2M,4M)  wq16   [4M,6M)  wv16
//   [6M,38M)  vk16   [38M,70M) vq16  [70M,102M) vv16
//   [102M,134M) kq16 (scores, then softmax in place)
// ---------------------------------------------------------------------------
extern "C" void kernel_launch(void* const* d_in, const int* in_sizes, int n_in,
                              void* d_out, int out_size, void* d_ws, size_t ws_size,
                              hipStream_t stream) {
    const float* x   = (const float*)d_in[0];
    const float* wkf = (const float*)d_in[1];
    const float* wqf = (const float*)d_in[2];
    const float* wvf = (const float*)d_in[3];
    float* out = (float*)d_out;

    char* ws = (char*)d_ws;
    const size_t MB = 1024ull * 1024ull;
    unsigned short* wk16 = (unsigned short*)(ws + 0 * MB);
    unsigned short* wq16 = (unsigned short*)(ws + 2 * MB);
    unsigned short* wv16 = (unsigned short*)(ws + 4 * MB);
    unsigned short* vk16 = (unsigned short*)(ws + 6 * MB);
    unsigned short* vq16 = (unsigned short*)(ws + 38 * MB);
    unsigned short* vv16 = (unsigned short*)(ws + 70 * MB);
    unsigned short* kq16 = (unsigned short*)(ws + 102 * MB);

    const int NW = 1024 * 1024;
    cvt_bf16_kernel<<<1024, 256, 0, stream>>>(wkf, wk16, NW);
    cvt_bf16_kernel<<<1024, 256, 0, stream>>>(wqf, wq16, NW);
    cvt_bf16_kernel<<<1024, 256, 0, stream>>>(wvf, wv16, NW);

    qkv_gemm_kernel<<<dim3(8, 8, 48), 256, 0, stream>>>(
        x, wk16, wq16, wv16, vk16, vq16, vv16);

    kq_gemm_kernel<<<dim3(8, 8, 16), 256, 0, stream>>>(vk16, vq16, kq16);

    softmax_kernel<<<16 * 1024, 256, 0, stream>>>(kq16);

    out_gemm_kernel<<<dim3(8, 8, 16), 256, 0, stream>>>(kq16, vv16, out);
}
